// ComplexRNNLayer_35811437314267
// MI455X (gfx1250) — matrix operand order repaired and checked by reading, not compiled, not run on hardware
//
#include <hip/hip_runtime.h>
#include <hip/hip_bf16.h>
#include <math.h>

// ---------------------------------------------------------------------------
// ComplexRNNLayer for MI455X (gfx1250, wave32, WMMA).
//
// Sizes: B=32, T=1024, H=512.  BT=32768, BTH=16,777,216 floats per output.
//
// Phase 1 (throughput):  x_r/x_i input projections as one WMMA f16 GEMM pass
//   (4 logical GEMMs fused per tile), results written into d_out (f32).
// Phase 2 (latency):     persistent scan kernel, 64 single-wave workgroups.
//   Each wave owns BOTH complex products for one (mt,nt) tile: the imag
//   product's A operand is a K-halves permutation of the real one, so each
//   A tile is loaded once and fed to two WMMAs (B0[kt], B1[kt^16]).
//   Recurrent weights (complex combination + sign folded into a concatenated
//   K=1024 B-matrix) resident in 64 KB LDS; hidden state ping-pong in
//   workspace (f16, L2-hot); one atomic grid barrier per timestep.
// ---------------------------------------------------------------------------

typedef __attribute__((ext_vector_type(16))) _Float16 v16h;
typedef __attribute__((ext_vector_type(8)))  float    v8f;

#define NB 32
#define NT 1024
#define NH 512
#define BT 32768                    // NB*NT
static const size_t BTH = (size_t)NB * NT * NH;

// ---- workspace layout (bytes). total ~3.2 MB -------------------------------
#define BAR_OFF   0u                 // 2 x u32 (count, generation)
#define BCAT_OFF  256u               // 1024 f32: [q=0: b_hr-b_hi | q=1: b_hr+b_hi]
#define H0_OFF    8192u              // 32*1024 f16 ping   (65536 B)
#define H1_OFF    (H0_OFF + 65536u)  // 32*1024 f16 pong
#define WREC_OFF  (H1_OFF + 65536u)  // packed recurrent B: 2*32*32*512 f16 = 2 MB
#define WIN_OFF   (WREC_OFF + 2097152u) // packed input B: 2*32*16*512 f16 = 1 MB

// ---------------------------------------------------------------------------
// WMMA operand layout helpers (CDNA5 ISA 7.12.2, wave32)
//  A (16x32 f16):  lane L: M = L&15; VGPR v holds K = (L>>4)*16
//                  + 2*v + {0,1}  -> VGPRs 0-3 and 4-7 are each one
//                  contiguous 16B run -> two b128 loads per A operand.
//  B (32x16 f16):  packed blob is consumed linearly: lane*16 + v*2 + h, with
//                  kk = (lane>>4)*16 + 2*v + h, col = lane&15 (pack == read).
//  C/D (16x16 f32): lane L: N = L&15; VGPR v holds M = v + 8*(L>>4).
// ---------------------------------------------------------------------------

__device__ __forceinline__ v16h load_a_from_f32(const float* rowp, int hi) {
  v16h a;
#pragma unroll
  for (int v = 0; v < 8; ++v) {
    const int kk = hi * 16 + 2 * v;
    const float2 f = *(const float2*)(rowp + kk);     // 8B-aligned
    a[2 * v]     = (_Float16)f.x;
    a[2 * v + 1] = (_Float16)f.y;
  }
  return a;
}

__device__ __forceinline__ v16h load_a_from_f16(const _Float16* rowp, int hi) {
  union { v16h h; unsigned u[8]; } a;
#pragma unroll
  for (int v = 0; v < 8; ++v) {
    const int kk = hi * 16 + 2 * v;
    a.u[v] = *(const unsigned*)(rowp + kk);           // merges into 2x b128
  }
  return a.h;
}

__device__ __forceinline__ v16h load_b_packed(const _Float16* p) { // 32B-aligned
  union { v16h h; float4 f[2]; } u;
  const float4* q = (const float4*)p;
  u.f[0] = q[0];
  u.f[1] = q[1];
  return u.h;
}

// ---------------------------------------------------------------------------
// init: zero barrier + hidden-state ping/pong, fold recurrent biases.
// grid 64 x 256 (tid covers 16384 uints per h buffer)
// ---------------------------------------------------------------------------
__global__ void crnn_init(unsigned* __restrict__ bar, float* __restrict__ bcat,
                          unsigned* __restrict__ h0u, unsigned* __restrict__ h1u,
                          const float* __restrict__ b_hr,
                          const float* __restrict__ b_hi) {
  const int tid = blockIdx.x * blockDim.x + threadIdx.x;
  if (tid < 16384) { h0u[tid] = 0u; h1u[tid] = 0u; }
  if (tid < NH) {
    const float r = b_hr[tid], i = b_hi[tid];
    bcat[tid]      = r - i;   // q=0 (real path)
    bcat[NH + tid] = r + i;   // q=1 (imag path)
  }
  if (tid == 0) { bar[0] = 0u; bar[1] = 0u; }
}

// ---------------------------------------------------------------------------
// pack recurrent weights into WMMA-B layout with complex combination folded:
//   q=0 (r_new):  K 0..511 -> W_hr[n][k],  K 512..1023 -> -W_hi[n][k-512]
//   q=1 (i_new):  K 0..511 -> W_hr[n][k],  K 512..1023 -> +W_hi[n][k-512]
// flat e (2^20): h(1) v(3) lane(5) kt(5) nt(5) q(1) -> blob index == e.
// ---------------------------------------------------------------------------
__global__ void crnn_pack_rec(_Float16* __restrict__ wrec,
                              const float* __restrict__ W_hr,
                              const float* __restrict__ W_hi) {
  const unsigned e = blockIdx.x * blockDim.x + threadIdx.x;  // < 1,048,576
  const int h    = e & 1;
  const int v    = (e >> 1) & 7;
  const int lane = (e >> 4) & 31;
  const int kt   = (e >> 9) & 31;
  const int nt   = (e >> 14) & 31;
  const int q    = (e >> 19) & 1;
  const int n  = nt * 16 + (lane & 15);
  const int kk = (lane >> 4) * 16 + 2 * v + h;
  const int kg = kt * 32 + kk;          // 0..1023
  float w;
  if (kg < NH) w = W_hr[n * NH + kg];
  else {
    w = W_hi[n * NH + (kg - NH)];
    if (q == 0) w = -w;
  }
  wrec[e] = (_Float16)w;
}

// flat e (2^19): h(1) v(3) lane(5) kt(4) nt(5) w(1) -> blob index == e.
__global__ void crnn_pack_in(_Float16* __restrict__ win,
                             const float* __restrict__ W_ir,
                             const float* __restrict__ W_ii) {
  const unsigned e = blockIdx.x * blockDim.x + threadIdx.x;  // < 524,288
  const int h    = e & 1;
  const int v    = (e >> 1) & 7;
  const int lane = (e >> 4) & 31;
  const int kt   = (e >> 9) & 15;
  const int nt   = (e >> 13) & 31;
  const int wsel = (e >> 18) & 1;
  const int n  = nt * 16 + (lane & 15);
  const int kk = (lane >> 4) * 16 + 2 * v + h;
  const float* W = wsel ? W_ii : W_ir;
  win[e] = (_Float16)W[n * NH + kt * 32 + kk];
}

// ---------------------------------------------------------------------------
// Phase 1: input projections.
//   x_r = r@Wir^T + b_ir - (i@Wii^T + b_ii)
//   x_i = i@Wir^T + b_ir + (r@Wii^T + b_ii)
// One wave per 16x16 output tile pair. grid 8192 x 256 (8 waves/block).
// Results land in d_out (f32), consumed in place by the scan.
// ---------------------------------------------------------------------------
__global__ __launch_bounds__(256) void crnn_xproj(
    const float* __restrict__ r_seq, const float* __restrict__ i_seq,
    const _Float16* __restrict__ win,
    const float* __restrict__ b_ir, const float* __restrict__ b_ii,
    float* __restrict__ out) {
  const int gtid = blockIdx.x * blockDim.x + threadIdx.x;
  const int wave = gtid >> 5;
  const int lane = threadIdx.x & 31;
  if (wave >= (BT / 16) * (NH / 16)) return;
  const int nt = wave & 31;          // 32 column tiles
  const int mt = wave >> 5;          // 2048 row tiles over B*T

  const int col = lane & 15;
  const int hi  = lane >> 4;
  const int row = mt * 16 + col;     // A-operand row for this lane

  v8f a_rWir = {}, a_iWir = {}, a_rWii = {}, a_iWii = {};

#pragma unroll 2
  for (int kt = 0; kt < 16; ++kt) {
    const float* rp = r_seq + (size_t)row * NH + kt * 32;
    const float* ip = i_seq + (size_t)row * NH + kt * 32;
    const v16h ar = load_a_from_f32(rp, hi);
    const v16h ai = load_a_from_f32(ip, hi);
    const _Float16* bir_p = win + ((size_t)((0 * 32 + nt) * 16 + kt)) * 512 + lane * 16;
    const _Float16* bii_p = win + ((size_t)((1 * 32 + nt) * 16 + kt)) * 512 + lane * 16;
    const v16h bir = load_b_packed(bir_p);
    const v16h bii = load_b_packed(bii_p);
    a_rWir = __builtin_amdgcn_wmma_f32_16x16x32_f16(false, ar, false, bir, (short)0, a_rWir, false, false);
    a_iWir = __builtin_amdgcn_wmma_f32_16x16x32_f16(false, ai, false, bir, (short)0, a_iWir, false, false);
    a_rWii = __builtin_amdgcn_wmma_f32_16x16x32_f16(false, ar, false, bii, (short)0, a_rWii, false, false);
    a_iWii = __builtin_amdgcn_wmma_f32_16x16x32_f16(false, ai, false, bii, (short)0, a_iWii, false, false);
  }

  const int n = nt * 16 + col;
  const float bir_v = b_ir[n];
  const float bii_v = b_ii[n];
#pragma unroll
  for (int v = 0; v < 8; ++v) {
    const int rg = mt * 16 + v + 8 * hi;             // global row in [B*T]
    const size_t off = (size_t)rg * NH + n;          // == (b*T + t)*H + n
    out[off]       = a_rWir[v] + bir_v - a_iWii[v] - bii_v;  // x_r
    out[BTH + off] = a_iWir[v] + bir_v + a_rWii[v] + bii_v;  // x_i
  }
}

// ---------------------------------------------------------------------------
// grid barrier: sense-counter in workspace, s_sleep spin.
// ---------------------------------------------------------------------------
__device__ __forceinline__ void grid_sync(unsigned* bar, unsigned nblk, unsigned step) {
  __threadfence();
  if (threadIdx.x == 0) {
    const unsigned old = __hip_atomic_fetch_add(&bar[0], 1u, __ATOMIC_ACQ_REL,
                                                __HIP_MEMORY_SCOPE_AGENT);
    if (old == nblk - 1u) {
      __hip_atomic_store(&bar[0], 0u, __ATOMIC_RELAXED, __HIP_MEMORY_SCOPE_AGENT);
      __hip_atomic_fetch_add(&bar[1], 1u, __ATOMIC_RELEASE, __HIP_MEMORY_SCOPE_AGENT);
    }
  }
  while (__hip_atomic_load(&bar[1], __ATOMIC_ACQUIRE, __HIP_MEMORY_SCOPE_AGENT) <= step)
    __builtin_amdgcn_s_sleep(1);
  __threadfence();
}

// ---------------------------------------------------------------------------
// Phase 2: persistent scan. 64 workgroups x 1 wave.
//   wg -> (nt = wg&31, mt = (wg>>5)&1); wave owns BOTH q products.
// Per step: 32 A-tile loads (h_cat rows mt, f16, L2-hot), 64 WMMAs
//   (acc0 += A[kt]*B0[kt], acc1 += A[kt]*B1[kt^16]).
// Both q weight slices resident in 64 KB LDS (of 320 KB/WGP).
// ---------------------------------------------------------------------------
__global__ __launch_bounds__(32) void crnn_scan(
    const float* __restrict__ r_seq, const float* __restrict__ i_seq,
    const _Float16* __restrict__ wrec, const float* __restrict__ bcat,
    _Float16* __restrict__ h0, _Float16* __restrict__ h1,
    float* __restrict__ out, unsigned* __restrict__ bar) {
  __shared__ __align__(16) _Float16 w_lds[2][32 * 512];   // 64 KB

  const int lane = threadIdx.x;
  const int wg   = blockIdx.x;
  const int nt   = wg & 31;
  const int mt   = (wg >> 5) & 1;

  // one-time LDS fill of this WG's B-slices (2 products x 32 kt x 512 halves)
  {
    const float4* s0 = (const float4*)(wrec + (size_t)((0 * 32 + nt) * 32) * 512);
    const float4* s1 = (const float4*)(wrec + (size_t)((1 * 32 + nt) * 32) * 512);
    float4* d0 = (float4*)w_lds[0];
    float4* d1 = (float4*)w_lds[1];
    for (int i = lane; i < 2048; i += 32) { d0[i] = s0[i]; d1[i] = s1[i]; }
  }
  __syncthreads();

  const int col = lane & 15;
  const int hi  = lane >> 4;
  const int n   = nt * 16 + col;
  const float bc0 = bcat[n];
  const float bc1 = bcat[NH + n];
  const unsigned nblk = gridDim.x;

  for (int t = 0; t < NT; ++t) {
    const _Float16* hcur = (t & 1) ? h1 : h0;
    _Float16* hnext      = (t & 1) ? h0 : h1;

    v8f acc0 = {}, acc1 = {};
#pragma unroll 2
    for (int kt = 0; kt < 32; ++kt) {
      const _Float16* hrow = hcur + (size_t)(mt * 16 + col) * 1024 + kt * 32;
      const v16h A  = load_a_from_f16(hrow, hi);
      const v16h B0 = load_b_packed(w_lds[0] + kt * 512 + lane * 16);
      const v16h B1 = load_b_packed(w_lds[1] + (kt ^ 16) * 512 + lane * 16);
      acc0 = __builtin_amdgcn_wmma_f32_16x16x32_f16(false, A, false, B0, (short)0, acc0, false, false);
      acc1 = __builtin_amdgcn_wmma_f32_16x16x32_f16(false, A, false, B1, (short)0, acc1, false, false);
    }

#pragma unroll
    for (int v = 0; v < 8; ++v) {
      const int b = mt * 16 + v + 8 * hi;
      const size_t off = ((size_t)b * NT + t) * NH + n;
      // real product
      {
        const float x = out[off];                    // precomputed x_r[b,t,n]
        const float hval = tanhf(acc0[v] + x + bc0);
        hnext[(size_t)b * 1024 + n] = (_Float16)hval;
        out[off] = hval + r_seq[off];                // residual, in-place over x
      }
      // imag product
      {
        const float x = out[BTH + off];              // precomputed x_i[b,t,n]
        const float hval = tanhf(acc1[v] + x + bc1);
        hnext[(size_t)b * 1024 + NH + n] = (_Float16)hval;
        out[BTH + off] = hval + i_seq[off];
      }
    }

    if (t + 1 < NT) {  // warm next timestep's x (global_prefetch_b8)
      const size_t poff = ((size_t)(mt * 16 + col) * NT + (t + 1)) * NH + n;
      __builtin_prefetch(&out[poff], 0, 3);
      __builtin_prefetch(&out[BTH + poff], 0, 3);
    }

    grid_sync(bar, nblk, (unsigned)t);
  }
}

// ---------------------------------------------------------------------------
extern "C" void kernel_launch(void* const* d_in, const int* in_sizes, int n_in,
                              void* d_out, int out_size, void* d_ws, size_t ws_size,
                              hipStream_t stream) {
  const float* r_seq = (const float*)d_in[0];
  const float* i_seq = (const float*)d_in[1];
  const float* W_ir  = (const float*)d_in[2];
  const float* b_ir  = (const float*)d_in[3];
  const float* W_ii  = (const float*)d_in[4];
  const float* b_ii  = (const float*)d_in[5];
  const float* W_hr  = (const float*)d_in[6];
  const float* b_hr  = (const float*)d_in[7];
  const float* W_hi  = (const float*)d_in[8];
  const float* b_hi  = (const float*)d_in[9];
  float* out = (float*)d_out;

  char* ws = (char*)d_ws;
  unsigned* bar   = (unsigned*)(ws + BAR_OFF);
  float*    bcat  = (float*)(ws + BCAT_OFF);
  _Float16* h0    = (_Float16*)(ws + H0_OFF);
  _Float16* h1    = (_Float16*)(ws + H1_OFF);
  _Float16* wrec  = (_Float16*)(ws + WREC_OFF);
  _Float16* win   = (_Float16*)(ws + WIN_OFF);

  crnn_init<<<64, 256, 0, stream>>>(bar, bcat, (unsigned*)h0, (unsigned*)h1, b_hr, b_hi);
  crnn_pack_rec<<<4096, 256, 0, stream>>>(wrec, W_hr, W_hi);
  crnn_pack_in<<<2048, 256, 0, stream>>>(win, W_ir, W_ii);
  crnn_xproj<<<8192, 256, 0, stream>>>(r_seq, i_seq, win, b_ir, b_ii, out);
  crnn_scan<<<64, 32, 0, stream>>>(r_seq, i_seq, wrec, bcat, h0, h1, out, bar);
}
